// TTConv_48326972015263
// MI455X (gfx1250) — matrix operand order repaired and checked by reading, not compiled, not run on hardware
//
#include <hip/hip_runtime.h>
#include <hip/hip_bf16.h>
#include <stdint.h>

// ---------------------------------------------------------------------------
// TT-conv for MI455X (gfx1250, wave32, WMMA).
//   out(B,512,32,32) = DWConv3x3(input) [as (M=32768, K=1024) bf16]
//                      x  chain-weights [as Wt (N=512, K=1024) bf16]
//                      + bias
// GEMM: v_wmma_f32_16x16x32_bf16 (f32 accum), triple-buffered LDS fed by
// GLOBAL_LOAD_ASYNC_TO_LDS_B128 (ASYNCcnt), one workgroup barrier per k-step.
// ---------------------------------------------------------------------------

typedef __attribute__((ext_vector_type(16))) __bf16 v16bf;
typedef __attribute__((ext_vector_type(8)))  float  v8f;
typedef __attribute__((ext_vector_type(4)))  int    v4i;

union FragAB { v16bf v; uint4 q[2]; };

#define BM 128
#define BN 128
#define BK 32
#define KDIM 1024
#define NDIM 512
#define MDIM 32768   // 32 batches * 32*32 spatial
#define KTILES (KDIM / BK)

#define AS1 __attribute__((address_space(1)))
#define AS3 __attribute__((address_space(3)))

#ifndef __has_builtin
#define __has_builtin(x) 0
#endif

#if __has_builtin(__builtin_amdgcn_global_load_async_to_lds_b128)
#define HAVE_ASYNC_LDS 1
// signature (probe-derived): (v4i AS1* src, v4i AS3* dst, imm offset, imm cpol)
#define ASYNC_B128(gsrc, ldst)                                            \
    __builtin_amdgcn_global_load_async_to_lds_b128(                       \
        (AS1 v4i*)(AS1 void*)(gsrc), (AS3 v4i*)(AS3 void*)(ldst), 0, 0)
#else
#define HAVE_ASYNC_LDS 0
#endif

#if __has_builtin(__builtin_amdgcn_s_wait_asynccnt)
#define WAIT_ASYNCCNT(n) __builtin_amdgcn_s_wait_asynccnt(n)
#else
#define WAIT_ASYNCCNT(n) asm volatile("s_wait_asynccnt %0" ::"n"(n) : "memory")
#endif

// ---------------------------------------------------------------------------
// Kernel 0: contract the TT cores into Wt[n][k] (bf16), n = x*64+y*8+z,
// k = a*128 + b*16 + c*4 + p.  Wt is stored N-major, K-contiguous so the GEMM
// B-tile fill is fully coalesced.
// ---------------------------------------------------------------------------
__global__ __launch_bounds__(256)
void tt_chain_kernel(const float* __restrict__ f0,   // (8,4,16,8)
                     const float* __restrict__ f1,   // (8,16,16,8)
                     const float* __restrict__ lf,   // (4,16,8)
                     __bf16* __restrict__ Wt)        // (512,1024)
{
    int idx = blockIdx.x * 256 + threadIdx.x;        // 0 .. 524287
    int k = idx & (KDIM - 1);
    int n = idx >> 10;
    int p = k & 3;
    int c = (k >> 2) & 3;
    int b = (k >> 4) & 7;
    int a = k >> 7;
    int z = n & 7;
    int y = (n >> 3) & 7;
    int x = n >> 6;

    float s = 0.0f;
    #pragma unroll
    for (int q = 0; q < 16; ++q) {
        float f0v = f0[((a * 4 + p) * 16 + q) * 8 + x];
        float inner = 0.0f;
        #pragma unroll
        for (int r = 0; r < 16; ++r)
            inner += f1[((b * 16 + q) * 16 + r) * 8 + y] * lf[(c * 16 + r) * 8 + z];
        s += f0v * inner;
    }
    Wt[(size_t)n * KDIM + k] = (__bf16)s;
}

// ---------------------------------------------------------------------------
// Kernel 1: depthwise 3x3 conv (pad 1), 4 shared filters per channel.
// One block per (batch, channel) plane. Output Y is (M x K) row-major bf16,
// M = b*1024 + h*32 + w, K = c*4 + p.
// ---------------------------------------------------------------------------
__global__ __launch_bounds__(256)
void tt_dwconv_kernel(const float* __restrict__ in,   // (32,256,32,32)
                      const float* __restrict__ sf,   // (4,1,3,3)
                      __bf16* __restrict__ Y)         // (32768,1024)
{
    __shared__ float tile[32][33];                    // +1 pad: no bank conflicts

    int bc = blockIdx.x;                              // 0 .. 8191
    int b  = bc >> 8;
    int c  = bc & 255;
    int tid = threadIdx.x;

    const float* plane = in + ((size_t)(b * 256 + c)) * 1024;
    #pragma unroll
    for (int t = 0; t < 4; ++t) {
        int pos = tid + t * 256;
        tile[pos >> 5][pos & 31] = plane[pos];
    }

    float w[4][9];
    #pragma unroll
    for (int p = 0; p < 4; ++p)
        #pragma unroll
        for (int i = 0; i < 9; ++i)
            w[p][i] = sf[p * 9 + i];

    __syncthreads();

    #pragma unroll
    for (int t = 0; t < 4; ++t) {
        int pos = tid * 4 + t;                        // 4 consecutive positions
        int h = pos >> 5;
        int wc = pos & 31;

        float acc[4] = {0.f, 0.f, 0.f, 0.f};
        #pragma unroll
        for (int dh = -1; dh <= 1; ++dh) {
            #pragma unroll
            for (int dw = -1; dw <= 1; ++dw) {
                int hh = h + dh, ww = wc + dw;
                float v = (hh >= 0 && hh < 32 && ww >= 0 && ww < 32)
                              ? tile[hh][ww] : 0.0f;
                int fi = (dh + 1) * 3 + (dw + 1);
                #pragma unroll
                for (int p = 0; p < 4; ++p) acc[p] += v * w[p][fi];
            }
        }

        union { __bf16 h4[4]; uint2 u; } pk;
        #pragma unroll
        for (int p = 0; p < 4; ++p) pk.h4[p] = (__bf16)acc[p];

        size_t m = (size_t)b * 1024 + pos;
        *(uint2*)(Y + m * KDIM + c * 4) = pk.u;       // 8B aligned packed store
    }
}

// ---------------------------------------------------------------------------
// WMMA inner step.
//  A 16x32 (bf16): m = lane&15, VGPR0-3 <- K[ak..ak+7], VGPR4-7 <- K[ak+16..+23],
//                  ak = (lane>=16)*8.
//  B 32x16 (bf16): n = lane&15, VGPR0-3 <- K[bk..bk+7], VGPR4-7 <- K[bk+16..+23],
//                  bk = (lane>=16)*8.
// ---------------------------------------------------------------------------
__device__ __forceinline__
void mma_step(const __bf16 (*Asb)[BK], const __bf16 (*Bsb)[BK],
              int wm, int wn, int lane, v8f acc[2][4])
{
    const int fam = lane & 15;
    const int fak = (lane >> 4) * 8;
    const int fbn = lane & 15;
    const int fbk = (lane >> 4) * 8;

    FragAB af[2], bfr[4];
    #pragma unroll
    for (int i = 0; i < 2; ++i) {
        int row = wm * 32 + i * 16 + fam;
        af[i].q[0] = *(const uint4*)&Asb[row][fak];
        af[i].q[1] = *(const uint4*)&Asb[row][fak + 16];
    }
    #pragma unroll
    for (int j = 0; j < 4; ++j) {
        int col = wn * 64 + j * 16 + fbn;
        bfr[j].q[0] = *(const uint4*)&Bsb[col][fbk];
        bfr[j].q[1] = *(const uint4*)&Bsb[col][fbk + 16];
    }
    #pragma unroll
    for (int i = 0; i < 2; ++i)
        #pragma unroll
        for (int j = 0; j < 4; ++j)
            acc[i][j] = __builtin_amdgcn_wmma_f32_16x16x32_bf16(
                false, af[i].v, false, bfr[j].v,
                (short)0, acc[i][j], false, false);
}

// ---------------------------------------------------------------------------
// Kernel 2: GEMM  out = Y(MxK) * Wt(NxK)^T + bias, with transposed NCHW store.
// 8 wave32 waves arranged 4(M) x 2(N); each wave owns a 32x64 sub-tile.
//
// Triple-buffered async pipeline, ONE barrier per k-step:
//   iter kt:  wait(tile kt landed) ; barrier ; issue async(tile kt+2) ;
//             compute tile kt.
// Safety: the buffer re-targeted by tile kt+2 last held tile kt-1, whose
// readers all finished BEFORE signaling barrier(kt); the write is issued
// AFTER barrier(kt) -> no read/write overlap with reuse distance 3.
// ---------------------------------------------------------------------------
__global__ __launch_bounds__(256)
void tt_gemm_kernel(const __bf16* __restrict__ A,    // (32768,1024) row-major
                    const __bf16* __restrict__ Bt,   // (512,1024)  N-major
                    const float* __restrict__ bias,  // (512)
                    float* __restrict__ out)         // (32,512,32,32)
{
#if HAVE_ASYNC_LDS
    __shared__ __bf16 As[3][BM][BK];                 // 3 x 8 KB
    __shared__ __bf16 Bs[3][BN][BK];                 // 3 x 8 KB
#else
    __shared__ __bf16 As[1][BM][BK];
    __shared__ __bf16 Bs[1][BN][BK];
#endif

    const int tid  = threadIdx.x;
    const int lane = tid & 31;
    const int wave = tid >> 5;
    const int wm   = wave & 3;                       // 4 waves along M
    const int wn   = wave >> 2;                      // 2 waves along N
    const int bm   = blockIdx.y * BM;
    const int bn   = blockIdx.x * BN;

    // Tile staging: 128x32 bf16 = 8 KB; each thread moves one 16 B chunk
    // per matrix per k-step.
    const int grow = tid >> 1;                       // 0..127
    const int gko  = (tid & 1) * 16;                 // 0 or 16 (bf16 units)

    const __bf16* gA = A  + (size_t)(bm + grow) * KDIM + gko;
    const __bf16* gB = Bt + (size_t)(bn + grow) * KDIM + gko;

    v8f acc[2][4];
    #pragma unroll
    for (int i = 0; i < 2; ++i)
        #pragma unroll
        for (int j = 0; j < 4; ++j) acc[i][j] = (v8f)0.0f;

#if HAVE_ASYNC_LDS
    // Prologue: tiles 0 and 1 in flight (2 async issues each per wave).
    ASYNC_B128(gA,      &As[0][grow][gko]);
    ASYNC_B128(gB,      &Bs[0][grow][gko]);
    ASYNC_B128(gA + BK, &As[1][grow][gko]);
    ASYNC_B128(gB + BK, &Bs[1][grow][gko]);

    int cur = 0;          // buffer holding tile kt
    int nx2 = 2;          // buffer receiving tile kt+2
    #pragma unroll 1
    for (int kt = 0; kt < KTILES; ++kt) {
        if (kt + 1 < KTILES) {
            WAIT_ASYNCCNT(2);                        // tile kt landed; kt+1 may fly
        } else {
            WAIT_ASYNCCNT(0);
        }
        __syncthreads();                             // tile kt resident everywhere;
                                                     // also: all reads of buf nx2
                                                     // (tile kt-1) are done.
        if (kt + 2 < KTILES) {
            ASYNC_B128(gA + (size_t)(kt + 2) * BK, &As[nx2][grow][gko]);
            ASYNC_B128(gB + (size_t)(kt + 2) * BK, &Bs[nx2][grow][gko]);
        }
        mma_step(As[cur], Bs[cur], wm, wn, lane, acc);
        cur = (cur == 2) ? 0 : cur + 1;
        nx2 = (nx2 == 2) ? 0 : nx2 + 1;
    }
#else
    // -------- fallback: synchronous global->VGPR->LDS staging -------------
    for (int k0 = 0; k0 < KDIM; k0 += BK) {
        const uint4 aReg = *(const uint4*)(gA + k0);
        const uint4 bReg = *(const uint4*)(gB + k0);
        if (k0 + BK < KDIM) {
            __builtin_prefetch(gA + k0 + BK, 0, 0);
            __builtin_prefetch(gB + k0 + BK, 0, 0);
        }
        __syncthreads();
        *(uint4*)&As[0][grow][gko] = aReg;
        *(uint4*)&Bs[0][grow][gko] = bReg;
        __syncthreads();
        mma_step(As[0], Bs[0], wm, wn, lane, acc);
    }
#endif

    // Epilogue. C/D layout: lane = (n_local | rowHalf<<4); VGPR v holds
    // M = tileM + rowHalf*8 + v for N = tileN + n_local -> the 8 accum floats
    // are 8 CONSECUTIVE hw positions of one output channel: two b128 stores
    // into the transposed (B,512,32,32) tensor.
    const int nl   = lane & 15;
    const int rofs = (lane >> 4) * 8;
    #pragma unroll
    for (int j = 0; j < 4; ++j) {
        const int n  = bn + wn * 64 + j * 16 + nl;
        const float bv = bias[n];
        #pragma unroll
        for (int i = 0; i < 2; ++i) {
            const int m  = bm + wm * 32 + i * 16 + rofs;  // tile stays in batch
            const int b  = m >> 10;
            const int hw = m & 1023;
            float* o = out + ((size_t)(b * NDIM + n)) * 1024 + hw;
            const float* s = (const float*)&acc[i][j];
            float4 r0, r1;
            r0.x = s[0] + bv; r0.y = s[1] + bv; r0.z = s[2] + bv; r0.w = s[3] + bv;
            r1.x = s[4] + bv; r1.y = s[5] + bv; r1.z = s[6] + bv; r1.w = s[7] + bv;
            *(float4*)(o)     = r0;
            *(float4*)(o + 4) = r1;
        }
    }
}

// ---------------------------------------------------------------------------
// Launch: chain -> dwconv -> gemm, all on `stream` (in-order dependency).
// Workspace layout: [0, 1MB)   Wt  (512*1024 bf16)
//                   [2MB, 66MB) Y  (32768*1024 bf16)
// ---------------------------------------------------------------------------
extern "C" void kernel_launch(void* const* d_in, const int* in_sizes, int n_in,
                              void* d_out, int out_size, void* d_ws, size_t ws_size,
                              hipStream_t stream)
{
    const float* input = (const float*)d_in[0];   // (32,256,32,32)
    const float* sf    = (const float*)d_in[1];   // (4,1,3,3)
    const float* f0    = (const float*)d_in[2];   // (8,4,16,8)
    const float* f1    = (const float*)d_in[3];   // (8,16,16,8)
    const float* lf    = (const float*)d_in[4];   // (4,16,8)
    const float* bias  = (const float*)d_in[5];   // (512)
    float* out         = (float*)d_out;           // (32,512,32,32)

    __bf16* Wt = (__bf16*)d_ws;                             // 1 MB
    __bf16* Y  = (__bf16*)((char*)d_ws + (2u << 20));       // 64 MB

    tt_chain_kernel<<<(NDIM * KDIM) / 256, 256, 0, stream>>>(f0, f1, lf, Wt);
    tt_dwconv_kernel<<<32 * 256, 256, 0, stream>>>(input, sf, Y);

    dim3 grid(NDIM / BN, MDIM / BM);   // (4, 256)
    tt_gemm_kernel<<<grid, 256, 0, stream>>>(Y, Wt, bias, out);
}
